// MultiQueryAttention_48095043780680
// MI455X (gfx1250) — compile-verified, matrix operand order checked by reading, not run on hardware
//
#include <hip/hip_runtime.h>
#include <hip/hip_bf16.h>

// ---------------------------------------------------------------------------
// MQA forward, all matmuls on v_wmma_f32_16x16x32_f16 (CDNA5 / gfx1250).
// B=2, S=2048, d_model=1024, H=16, Dh=64 (single shared K/V head).
// GEMMs: 4-wave blocks, B-operand staged in LDS via global_load_async_to_lds
// (double buffered, ASYNCcnt), A-operand software-pipelined in registers.
// Attention: 16-wave blocks (one head per wave) sharing async-staged,
// double-buffered K/V tiles in LDS -- 16x cut in L2 traffic for the shared
// MQA head.
// ---------------------------------------------------------------------------

typedef __attribute__((ext_vector_type(16))) _Float16 v16h;
typedef __attribute__((ext_vector_type(8)))  _Float16 h8;
typedef __attribute__((ext_vector_type(8)))  float    v8f;
typedef __attribute__((ext_vector_type(2)))  _Float16 h2;
typedef __attribute__((ext_vector_type(4)))  int      v4i;

#define BB    2
#define SS    2048
#define BSZ   4096      // B*S
#define DIN   1024
#define DHID  1024
#define NH    16
#define DH    64
#define SCALE 0.03125f  // 1/sqrt(1024)

#if __has_builtin(__builtin_amdgcn_global_load_async_to_lds_b128)
#define HAVE_ASYNC 1
#else
#define HAVE_ASYNC 0
#endif

// 16-byte global -> LDS copy; async (ASYNCcnt) when the gfx1250 builtin exists.
__device__ __forceinline__ void cp16(const _Float16* g, _Float16* l) {
#if HAVE_ASYNC
  __builtin_amdgcn_global_load_async_to_lds_b128(
      (__attribute__((address_space(1))) v4i*)g,
      (__attribute__((address_space(3))) v4i*)l, 0, 0);
#else
  *(h8*)l = *(const h8*)g;
#endif
}

// Wait until at most N async copies remain in flight.
template <int N>
__device__ __forceinline__ void wait_async() {
#if HAVE_ASYNC
#if __has_builtin(__builtin_amdgcn_s_wait_asynccnt)
  __builtin_amdgcn_s_wait_asynccnt(N);
#else
  asm volatile("s_wait_asynccnt %0" ::"i"(N) : "memory");
#endif
#endif
}

// Assemble a 16-bit WMMA A/B fragment (16x32 resp. 32x16) for one K=32 chunk.
// Per ISA 7.12.2: lane holds row/col (lane&15); hi = lane>>4 selects the K
// half-groups. Half t -> K = t + 8*hi + 8*(t>=8); pairs merge into b128 loads.
__device__ __forceinline__ v16h frag16(const _Float16* __restrict__ base, int hi) {
  v16h r;
#pragma unroll
  for (int tp = 0; tp < 8; ++tp) {
    const int kk = 2 * tp + 8 * hi + ((tp >= 4) ? 8 : 0);
    h2 p = *(const h2*)(base + kk);
    r[2 * tp]     = p.x;
    r[2 * tp + 1] = p.y;
  }
  return r;
}

// ---------------------------------------------------------------------------
// fp32 -> f16 casts (one-time, memory bound, ~3us of the budget)
// ---------------------------------------------------------------------------
__global__ void cast_h(const float* __restrict__ in, _Float16* __restrict__ out,
                       int n) {
  int i = blockIdx.x * blockDim.x + threadIdx.x;
  if (i < n) out[i] = (_Float16)in[i];
}

// W [K,N] fp32 row-major  ->  W^T [N,K] f16 row-major
__global__ void cast_t(const float* __restrict__ in, _Float16* __restrict__ out,
                       int K, int N) {
  int i = blockIdx.x * blockDim.x + threadIdx.x;
  if (i < K * N) {
    int k = i / N, n = i - k * N;
    out[(size_t)n * K + k] = (_Float16)in[i];
  }
}

// ---------------------------------------------------------------------------
// WMMA GEMM:  C[M,N] = X[M,K] @ W[K,N] + bias,  W given as WT[N,K].
// Block = 4 waves = 64x64 tile. B chunk (64x32 f16 = 4KB) staged in LDS by
// async copies, double buffered; each wave owns one 16-row strip and 4 f32
// accumulators; A fragments pipelined in registers.
// ---------------------------------------------------------------------------
__global__ void __launch_bounds__(128)
gemm_wmma(const _Float16* __restrict__ X,
          const _Float16* __restrict__ WT,
          const float* __restrict__ bias,
          _Float16* __restrict__ outH,
          float* __restrict__ outF,
          int M, int N, int K, int transposeOut) {
  __shared__ _Float16 bsh[2][64 * 32];   // [buf][row(64) x k(32)]

  const int tid  = threadIdx.x;          // 0..127
  const int wave = tid >> 5;
  const int lane = tid & 31;
  const int nl   = lane & 15;
  const int hi   = lane >> 4;
  const int m0   = (blockIdx.x * 4 + wave) * 16;
  const int n0   = blockIdx.y * 64;

  const _Float16* arow = X + (size_t)(m0 + nl) * K;

  // Stage one 64x32 B chunk: 256 x 16B segments, 2 per thread.
  auto stageB = [&](int k0, int buf) {
#pragma unroll
    for (int i = 0; i < 2; ++i) {
      const int seg = tid + i * 128;
      const int row = seg >> 2;          // 4 segs per 32-half row
      const int c16 = seg & 3;
      cp16(WT + (size_t)(n0 + row) * K + k0 + c16 * 8,
           &bsh[buf][row * 32 + c16 * 8]);
    }
  };

  v8f z = {};
  v8f c[4] = {z, z, z, z};

  stageB(0, 0);
  v16h a_cur = frag16(arow, hi);

  for (int k0 = 0; k0 < K; k0 += 32) {
    const int buf  = (k0 >> 5) & 1;
    const int nk0  = k0 + 32;
    const bool more = nk0 < K;
    if (more) { stageB(nk0, buf ^ 1); wait_async<2>(); }  // next copy in flight
    else      { wait_async<0>(); }
    __syncthreads();                     // B chunk visible to all 4 waves

    v16h a_nxt = more ? frag16(arow + nk0, hi) : a_cur;  // pipeline A
#pragma unroll
    for (int nt = 0; nt < 4; ++nt) {
      v16h b = frag16(&bsh[buf][(nt * 16 + nl) * 32], hi);
      c[nt] = __builtin_amdgcn_wmma_f32_16x16x32_f16(
          false, a_cur, false, b, (short)0, c[nt], false, false);
    }
    a_cur = a_nxt;
    __syncthreads();                     // before next stage overwrites buf^1
  }

#pragma unroll
  for (int nt = 0; nt < 4; ++nt) {
    const int gn = n0 + nt * 16 + nl;
    const float bval = bias ? bias[gn] : 0.0f;
#pragma unroll
    for (int r = 0; r < 8; ++r) {
      const int gm = m0 + r + 8 * hi;    // C/D layout: VGPR r -> row
      const float val = c[nt][r] + bval;
      if (outH) {
        if (transposeOut) outH[(size_t)gn * M + gm] = (_Float16)val;
        else              outH[(size_t)gm * N + gn] = (_Float16)val;
      } else {
        outF[(size_t)gm * N + gn] = val;
      }
    }
  }
}

// ---------------------------------------------------------------------------
// Flash-style MQA attention. Block = 16 waves = all 16 heads of one
// (batch, 16-query tile); the shared K/V head is staged once per block into
// LDS (async, double buffered) instead of once per wave.
//   qh : [B*S, 1024] f16    kh : [B*S, 64] f16    vT : [64, B*S] f16
//   oh : [B*S, 1024] f16 (attn output, pre O-projection)
// ---------------------------------------------------------------------------
__global__ void __launch_bounds__(512)
mqa_attn(const _Float16* __restrict__ qh,
         const _Float16* __restrict__ kh,
         const _Float16* __restrict__ vT,
         _Float16* __restrict__ oh) {
  __shared__ _Float16 ksh[2][32 * DH];    // [buf][key row(32) x d(64)]  8KB
  __shared__ _Float16 vsh[2][DH * 32];    // [buf][vcol(64) x key(32)]   8KB
  __shared__ _Float16 pl[NH][16 * 32];    // per-wave P staging          16KB

  const int tid  = threadIdx.x;           // 0..511
  const int wave = tid >> 5;              // = head
  const int lane = tid & 31;
  const int nl   = lane & 15;
  const int hi   = lane >> 4;
  const int qt   = blockIdx.x;            // query tile [0,128)
  const int b    = blockIdx.y;            // batch
  const int h    = wave;

  // Stage the next 32-key K tile + V^T tile: 512 x 16B segs, 1 per thread.
  auto stageKV = [&](int j, int buf) {
    if (tid < 256) {                      // K tile: 32 rows x 64 halves
      const int row = tid >> 3, s = tid & 7;
      cp16(kh + ((size_t)(b * SS + j + row)) * DH + s * 8,
           &ksh[buf][row * DH + s * 8]);
    } else {                              // V^T tile: 64 rows x 32 halves
      const int t2 = tid - 256;
      const int row = t2 >> 2, s = t2 & 3;
      cp16(vT + (size_t)row * BSZ + b * SS + j + s * 8,
           &vsh[buf][row * 32 + s * 8]);
    }
  };

  // Q tile resident as two A fragments (d = 0..31, 32..63)
  const _Float16* qrow = qh + ((size_t)(b * SS + qt * 16 + nl)) * DHID + h * DH;
  const v16h qa0 = frag16(qrow, hi);
  const v16h qa1 = frag16(qrow + 32, hi);

  v8f z = {};
  v8f o[4] = {z, z, z, z};                // O accum, 16x64
  float m8[8], l8[8];
#pragma unroll
  for (int r = 0; r < 8; ++r) { m8[r] = -3.0e38f; l8[r] = 0.0f; }

  stageKV(0, 0);

  for (int j = 0; j < SS; j += 32) {
    const int buf   = (j >> 5) & 1;
    const bool more = (j + 32) < SS;
    if (more) { stageKV(j + 32, buf ^ 1); wait_async<1>(); }
    else      { wait_async<0>(); }
    __syncthreads();                      // K/V tiles visible to all heads

    // ---- scores: two 16x16 tiles, S = Q @ K^T, B operand from LDS ----
    const _Float16* kr0 = &ksh[buf][nl * DH];
    const _Float16* kr1 = &ksh[buf][(16 + nl) * DH];
    v8f s0 = z, s1 = z;
    s0 = __builtin_amdgcn_wmma_f32_16x16x32_f16(false, qa0, false, frag16(kr0, hi),      (short)0, s0, false, false);
    s0 = __builtin_amdgcn_wmma_f32_16x16x32_f16(false, qa1, false, frag16(kr0 + 32, hi), (short)0, s0, false, false);
    s1 = __builtin_amdgcn_wmma_f32_16x16x32_f16(false, qa0, false, frag16(kr1, hi),      (short)0, s1, false, false);
    s1 = __builtin_amdgcn_wmma_f32_16x16x32_f16(false, qa1, false, frag16(kr1 + 32, hi), (short)0, s1, false, false);

    // ---- online softmax over the 16x32 slab ----
    float corr[8];
#pragma unroll
    for (int r = 0; r < 8; ++r) {
      const float x0 = s0[r] * SCALE;
      const float x1 = s1[r] * SCALE;
      float t = fmaxf(x0, x1);
      t = fmaxf(t, __shfl_xor(t, 1));
      t = fmaxf(t, __shfl_xor(t, 2));
      t = fmaxf(t, __shfl_xor(t, 4));
      t = fmaxf(t, __shfl_xor(t, 8));     // row max within 16-lane group
      const float mnew = fmaxf(m8[r], t);
      const float p0 = __expf(x0 - mnew);
      const float p1 = __expf(x1 - mnew);
      float rs = p0 + p1;
      rs += __shfl_xor(rs, 1);
      rs += __shfl_xor(rs, 2);
      rs += __shfl_xor(rs, 4);
      rs += __shfl_xor(rs, 8);            // row sum
      const float cr = __expf(m8[r] - mnew);
      l8[r] = l8[r] * cr + rs;
      m8[r] = mnew;
      corr[r] = cr;
      // C-layout: lane owns column nl of rows r+8*hi
      pl[wave][(r + 8 * hi) * 32 + nl]      = (_Float16)p0;
      pl[wave][(r + 8 * hi) * 32 + 16 + nl] = (_Float16)p1;
    }
#pragma unroll
    for (int r = 0; r < 8; ++r) {
      o[0][r] *= corr[r]; o[1][r] *= corr[r];
      o[2][r] *= corr[r]; o[3][r] *= corr[r];
    }

    // ---- P (A-layout from LDS) @ V (B operand from LDS V^T tile) ----
    const v16h pa = frag16(&pl[wave][nl * 32], hi);
#pragma unroll
    for (int nt = 0; nt < 4; ++nt) {
      v16h bv = frag16(&vsh[buf][(nt * 16 + nl) * 32], hi);
      o[nt] = __builtin_amdgcn_wmma_f32_16x16x32_f16(
          false, pa, false, bv, (short)0, o[nt], false, false);
    }
    __syncthreads();                      // before next stage overwrites buf^1
  }

  // ---- normalize and write attention output ----
#pragma unroll
  for (int r = 0; r < 8; ++r) {
    const float inv = 1.0f / l8[r];
    const int gm = qt * 16 + r + 8 * hi;
#pragma unroll
    for (int nt = 0; nt < 4; ++nt) {
      oh[((size_t)(b * SS + gm)) * DHID + h * DH + nt * 16 + nl] =
          (_Float16)(o[nt][r] * inv);
    }
  }
}

// ---------------------------------------------------------------------------
// Launcher
// ---------------------------------------------------------------------------
extern "C" void kernel_launch(void* const* d_in, const int* in_sizes, int n_in,
                              void* d_out, int out_size, void* d_ws, size_t ws_size,
                              hipStream_t stream) {
  (void)in_sizes; (void)n_in; (void)out_size; (void)ws_size;

  const float* query = (const float*)d_in[0];
  const float* key   = (const float*)d_in[1];
  const float* value = (const float*)d_in[2];
  const float* Wq    = (const float*)d_in[3];
  const float* bq    = (const float*)d_in[4];
  const float* Wk    = (const float*)d_in[5];
  const float* bk    = (const float*)d_in[6];
  const float* Wv    = (const float*)d_in[7];
  const float* bv    = (const float*)d_in[8];
  const float* Wo    = (const float*)d_in[9];
  const float* bo    = (const float*)d_in[10];
  float* outF = (float*)d_out;

  // workspace carve-up (f16, 256B aligned), ~45 MB total
  char* ws = (char*)d_ws;
  size_t off = 0;
  auto carve = [&](size_t halves) {
    void* p = ws + off;
    off += (halves * sizeof(_Float16) + 255) & ~(size_t)255;
    return (_Float16*)p;
  };
  _Float16* Xq  = carve((size_t)BSZ * DIN);
  _Float16* Xk  = carve((size_t)BSZ * DIN);
  _Float16* Xv  = carve((size_t)BSZ * DIN);
  _Float16* WqT = carve((size_t)DIN * DHID);
  _Float16* WkT = carve((size_t)DH * DIN);
  _Float16* WvT = carve((size_t)DH * DIN);
  _Float16* WoT = carve((size_t)DIN * DHID);
  _Float16* qp  = carve((size_t)BSZ * DHID);
  _Float16* kp  = carve((size_t)BSZ * DH);
  _Float16* vTp = carve((size_t)DH * BSZ);   // V^T [64, B*S]
  _Float16* oat = carve((size_t)BSZ * DHID);

  const int THREADS = 256;
  const int nAct = BSZ * DIN;
  cast_h<<<dim3((nAct + THREADS - 1) / THREADS), THREADS, 0, stream>>>(query, Xq, nAct);
  cast_h<<<dim3((nAct + THREADS - 1) / THREADS), THREADS, 0, stream>>>(key,   Xk, nAct);
  cast_h<<<dim3((nAct + THREADS - 1) / THREADS), THREADS, 0, stream>>>(value, Xv, nAct);

  cast_t<<<dim3((DIN * DHID + THREADS - 1) / THREADS), THREADS, 0, stream>>>(Wq, WqT, DIN, DHID);
  cast_t<<<dim3((DIN * DH   + THREADS - 1) / THREADS), THREADS, 0, stream>>>(Wk, WkT, DIN, DH);
  cast_t<<<dim3((DIN * DH   + THREADS - 1) / THREADS), THREADS, 0, stream>>>(Wv, WvT, DIN, DH);
  cast_t<<<dim3((DHID * DIN + THREADS - 1) / THREADS), THREADS, 0, stream>>>(Wo, WoT, DHID, DIN);

  // projections (WMMA, 64x64 tiles per 4-wave block)
  gemm_wmma<<<dim3(BSZ / 64, DHID / 64), 128, 0, stream>>>(Xq, WqT, bq, qp, nullptr, BSZ, DHID, DIN, 0);
  gemm_wmma<<<dim3(BSZ / 64, 1),         128, 0, stream>>>(Xk, WkT, bk, kp, nullptr, BSZ, DH,   DIN, 0);
  gemm_wmma<<<dim3(BSZ / 64, 1),         128, 0, stream>>>(Xv, WvT, bv, vTp, nullptr, BSZ, DH,  DIN, 1);

  // attention (WMMA flash, 16 heads per block sharing LDS-staged K/V)
  mqa_attn<<<dim3(SS / 16, BB), 512, 0, stream>>>(qp, kp, vTp, oat);

  // output projection (WMMA, fp32 out)
  gemm_wmma<<<dim3(BSZ / 64, DIN / 64), 128, 0, stream>>>(oat, WoT, bo, nullptr, outF, BSZ, DIN, DHID, 0);
}
